// MultiHeadAttention_59665685676363
// MI455X (gfx1250) — matrix operand order repaired and checked by reading, not compiled, not run on hardware
//
#include <hip/hip_runtime.h>

// Differential multi-head attention for MI455X (gfx1250, wave32, WMMA).
// Pipeline: qkv projection (WMMA, layout-permuting stores) -> flash-style
// dual-softmax attention (WMMA + async-to-LDS staging + s_wait_* + prefetch)
// -> group-norm stats -> normalized gather + output projection (WMMA).

#define B_ 4
#define H_ 8
#define S_ 1024
#define D_ 128
#define SCALE 0.08838834764831845f  // 1/sqrt(128)

typedef _Float16 f16;
typedef __attribute__((ext_vector_type(16))) _Float16 v16h;
typedef __attribute__((ext_vector_type(8)))  _Float16 v8h;
typedef __attribute__((ext_vector_type(8)))  float    v8f;
typedef int v4i_g __attribute__((vector_size(16)));  // matches builtin param

#define AS1 __attribute__((address_space(1)))
#define AS3 __attribute__((address_space(3)))

// CDNA5 async global->LDS staging (ASYNCcnt-tracked), if the toolchain has it.
#if __has_builtin(__builtin_amdgcn_global_load_async_to_lds_b128)
#define HAVE_ASYNC_LDS 1
#endif

static __device__ __forceinline__ void wait_asynccnt0() {
#if __has_builtin(__builtin_amdgcn_s_wait_asynccnt)
  __builtin_amdgcn_s_wait_asynccnt(0);
#else
  asm volatile("s_wait_asynccnt 0" ::: "memory");
#endif
}

// Issue one 16-byte async global->LDS copy (global_load_async_to_lds_b128).
static __device__ __forceinline__ void async_copy16(const void* gsrc,
                                                    void* ldst) {
#if defined(HAVE_ASYNC_LDS)
  __builtin_amdgcn_global_load_async_to_lds_b128(
      (AS1 v4i_g*)gsrc, (AS3 v4i_g*)ldst, 0, 0);
#else
  *(v4i_g*)ldst = *(const v4i_g*)gsrc;
#endif
}

static __device__ __forceinline__ v8f wmma_f16(v16h a, v16h b, v8f c) {
  // D = A(16x32 f16) * B(32x16 f16) + C(16x16 f32)
  return __builtin_amdgcn_wmma_f32_16x16x32_f16(false, a, false, b, (short)0, c,
                                                false, false);
}

// Build an A-fragment from row-major f16 memory: lane holds K = koff+{0..15}.
// Caller passes p already offset by koff.
static __device__ __forceinline__ v16h make_afrag(const f16* p) {
  v8h lo = *(const v8h*)p;
  v8h hi = *(const v8h*)(p + 8);
  v16h r;
#pragma unroll
  for (int i = 0; i < 8; i++) { r[i] = lo[i]; r[i + 8] = hi[i]; }
  return r;
}

// Online-softmax update for a 16x64 score block held as 4 D-layout v8f tiles.
// Row stats are replicated across the 16 lanes of each half-wave; xor masks
// < 16 keep the reduction within each half (rows 0-7 vs 8-15).
static __device__ __forceinline__ void online_update(v8f (&sacc)[4],
                                                     float (&m)[8],
                                                     float (&l)[8],
                                                     float (&osc)[8]) {
#pragma unroll
  for (int r = 0; r < 8; r++) {
    float v = fmaxf(fmaxf(sacc[0][r], sacc[1][r]), fmaxf(sacc[2][r], sacc[3][r]));
#pragma unroll
    for (int msk = 8; msk >= 1; msk >>= 1) v = fmaxf(v, __shfl_xor(v, msk, 32));
    float mn = fmaxf(m[r], v);
    float sc = __expf(m[r] - mn);
    m[r] = mn;
    float rs = 0.0f;
#pragma unroll
    for (int nt = 0; nt < 4; nt++) {
      float p = __expf(sacc[nt][r] - mn);
      sacc[nt][r] = p;
      rs += p;
    }
#pragma unroll
    for (int msk = 8; msk >= 1; msk >>= 1) rs += __shfl_xor(rs, msk, 32);
    l[r] = l[r] * sc + rs;
    osc[r] = sc;
  }
}

// ---------------------------------------------------------------------------
// Kernel 1: projection  out = X @ W^T + b  with layout-permuting f16 stores.
// mode 0: N=2048, split halves into out1/out2 (q1/q2 or k1/k2), per-head (S,128)
// mode 1: N=1024, store V transposed per head as (128, S)
// Grid: (M/64, N/64), block 128 (4 waves, 16 rows each).
// ---------------------------------------------------------------------------
__global__ __launch_bounds__(128, 1)
void proj_kernel(const float* __restrict__ X, const float* __restrict__ W,
                 const float* __restrict__ bias, f16* __restrict__ out1,
                 f16* __restrict__ out2, int mode) {
  const int tid = threadIdx.x, lane = tid & 31, w = tid >> 5;
  const int lhalf = lane >> 4, n16 = lane & 15;
  const int koffA = lhalf ? 16 : 0, koffB = lhalf ? 16 : 0;
  const int mtile = blockIdx.x * 64, ntile = blockIdx.y * 64;

  // A fragments: 16 rows x 128 K (f32 -> f16 on load)
  const float* xr = X + (size_t)(mtile + w * 16 + n16) * 128;
  v16h af[4];
#pragma unroll
  for (int ks = 0; ks < 4; ks++) {
    const float* p = xr + ks * 32 + koffA;
    v16h a;
#pragma unroll
    for (int i = 0; i < 8; i++) { a[i] = (f16)p[i]; a[i + 8] = (f16)p[8 + i]; }
    af[ks] = a;
  }

#pragma unroll
  for (int nt = 0; nt < 4; nt++) {
    const int ncol = ntile + nt * 16 + n16;
    const float* wr = W + (size_t)ncol * 128;
    v8f acc = {};
#pragma unroll
    for (int ks = 0; ks < 4; ks++) {
      const float* p = wr + ks * 32 + koffB;
      v16h bf;
#pragma unroll
      for (int i = 0; i < 16; i++) bf[i] = (f16)p[i];
      acc = wmma_f16(af[ks], bf, acc);
    }
    const float bval = bias[ncol];
#pragma unroll
    for (int r = 0; r < 8; r++) {
      const int m = mtile + w * 16 + r + 8 * lhalf;  // = b*1024 + s
      const float val = acc[r] + bval;
      const int b = m >> 10, s = m & 1023;
      const int hh = s >> 7, sl = s & 127;
      if (mode == 0) {
        const int s2 = (sl << 3) + (ncol >> 8);
        const int dd = ncol & 255;
        f16* dst = (dd < 128) ? out1 : out2;
        dst[((size_t)(b * 8 + hh) * 1024 + s2) * 128 + (dd & 127)] = (f16)val;
      } else {
        const int s2 = (sl << 3) + (ncol >> 7);
        const int dd = ncol & 127;
        out1[((size_t)(b * 8 + hh) * 128 + dd) * 1024 + s2] = (f16)val;
      }
    }
  }
}

// ---------------------------------------------------------------------------
// Kernel 2: dual-softmax flash attention per (b,h).
// Grid: (S/64, B*H), block 128 (4 waves, 16 query rows each).
// ---------------------------------------------------------------------------
__global__ __launch_bounds__(128, 1)
void attn_kernel(const f16* __restrict__ Q1, const f16* __restrict__ Q2,
                 const f16* __restrict__ K1, const f16* __restrict__ K2,
                 const f16* __restrict__ Vt, const float* __restrict__ lamP,
                 float* __restrict__ A) {
  const int tid = threadIdx.x, lane = tid & 31, w = tid >> 5;
  const int lhalf = lane >> 4, n16 = lane & 15;
  const int koffA = lhalf ? 16 : 0, koffB = lhalf ? 16 : 0;
  const int q0 = blockIdx.x * 64;
  const int bh = blockIdx.y;
  const float lam = lamP[0];
  const size_t base = (size_t)bh * (S_ * D_);
  const f16* gQ1 = Q1 + base;
  const f16* gQ2 = Q2 + base;
  const f16* gK1 = K1 + base;
  const f16* gK2 = K2 + base;
  const f16* gVt = Vt + base;  // (128, S) per head

  __shared__ __align__(32) f16 sK1[64][128];   // 16 KB
  __shared__ __align__(32) f16 sK2[64][128];   // 16 KB
  __shared__ __align__(32) f16 sVt[128][64];   // 16 KB
  __shared__ __align__(32) f16 sP[4][16][64];  // 8 KB (per-wave P scratch)

  // Query fragments (16 x 128, both branches)
  const f16* qr1 = gQ1 + (size_t)(q0 + w * 16 + n16) * 128;
  const f16* qr2 = gQ2 + (size_t)(q0 + w * 16 + n16) * 128;
  v16h q1f[4], q2f[4];
#pragma unroll
  for (int ks = 0; ks < 4; ks++) {
    q1f[ks] = make_afrag(qr1 + ks * 32 + koffA);
    q2f[ks] = make_afrag(qr2 + ks * 32 + koffA);
  }

  v8f zero = {};
  v8f O1[8], O2[8];
  float m1[8], l1[8], m2[8], l2[8];
#pragma unroll
  for (int i = 0; i < 8; i++) {
    O1[i] = zero; O2[i] = zero;
    m1[i] = -1e30f; m2[i] = -1e30f; l1[i] = 0.0f; l2[i] = 0.0f;
  }

  for (int kb = 0; kb < S_; kb += 64) {
    __syncthreads();  // previous tile fully consumed
    {   // cooperative async stage of K1/K2/Vt key-block tiles (ASYNCcnt)
      const uint4* s1p = (const uint4*)(gK1 + (size_t)kb * 128);
      const uint4* s2p = (const uint4*)(gK2 + (size_t)kb * 128);
      uint4* d1 = (uint4*)&sK1[0][0];
      uint4* d2 = (uint4*)&sK2[0][0];
      for (int i = tid; i < 1024; i += 128) {
        async_copy16(s1p + i, d1 + i);
        async_copy16(s2p + i, d2 + i);
      }
      for (int i = tid; i < 1024; i += 128) {
        const int dd = i >> 3, c = i & 7;
        async_copy16(gVt + (size_t)dd * S_ + kb + c * 8, &sVt[dd][c * 8]);
      }
#if defined(HAVE_ASYNC_LDS)
      wait_asynccnt0();
#endif
    }
    __syncthreads();
    if (kb + 64 < S_)  // prefetch next key block -> global_prefetch_b8
      __builtin_prefetch(gK1 + (size_t)(kb + 64) * 128 + tid * 64, 0, 1);

    // Scores: S1 = Q1*K1^T, S2 = Q2*K2^T (16 WMMA each)
    v8f s1acc[4], s2acc[4];
#pragma unroll
    for (int nt = 0; nt < 4; nt++) {
      s1acc[nt] = zero; s2acc[nt] = zero;
#pragma unroll
      for (int ks = 0; ks < 4; ks++) {
        v16h b1 = *(const v16h*)&sK1[nt * 16 + n16][ks * 32 + koffB];
        v16h b2 = *(const v16h*)&sK2[nt * 16 + n16][ks * 32 + koffB];
        s1acc[nt] = wmma_f16(q1f[ks], b1, s1acc[nt]);
        s2acc[nt] = wmma_f16(q2f[ks], b2, s2acc[nt]);
      }
    }
#pragma unroll
    for (int nt = 0; nt < 4; nt++) { s1acc[nt] *= SCALE; s2acc[nt] *= SCALE; }

    float osc1[8], osc2[8];
    online_update(s1acc, m1, l1, osc1);
    online_update(s2acc, m2, l2, osc2);
#pragma unroll
    for (int ot = 0; ot < 8; ot++)
#pragma unroll
      for (int r = 0; r < 8; r++) { O1[ot][r] *= osc1[r]; O2[ot][r] *= osc2[r]; }

    // P1: D-layout -> LDS -> A-layout fragments
#pragma unroll
    for (int nt = 0; nt < 4; nt++)
#pragma unroll
      for (int r = 0; r < 8; r++)
        sP[w][r + 8 * lhalf][nt * 16 + n16] = (f16)s1acc[nt][r];
    asm volatile("s_wait_dscnt 0" ::: "memory");
    v16h p1f[2];
#pragma unroll
    for (int pk = 0; pk < 2; pk++)
      p1f[pk] = make_afrag(&sP[w][n16][pk * 32 + koffA]);

    // P2 reuses the same per-wave scratch (DS in-order within a wave)
#pragma unroll
    for (int nt = 0; nt < 4; nt++)
#pragma unroll
      for (int r = 0; r < 8; r++)
        sP[w][r + 8 * lhalf][nt * 16 + n16] = (f16)s2acc[nt][r];
    asm volatile("s_wait_dscnt 0" ::: "memory");
    v16h p2f[2];
#pragma unroll
    for (int pk = 0; pk < 2; pk++)
      p2f[pk] = make_afrag(&sP[w][n16][pk * 32 + koffA]);

    // O += P * V  (B fragment from transposed V tile, shared by both branches)
#pragma unroll
    for (int ot = 0; ot < 8; ot++)
#pragma unroll
      for (int pk = 0; pk < 2; pk++) {
        v16h bv = *(const v16h*)&sVt[ot * 16 + n16][pk * 32 + koffB];
        O1[ot] = wmma_f16(p1f[pk], bv, O1[ot]);
        O2[ot] = wmma_f16(p2f[pk], bv, O2[ot]);
      }
  }

  float inv1[8], inv2[8];
#pragma unroll
  for (int r = 0; r < 8; r++) { inv1[r] = 1.0f / l1[r]; inv2[r] = lam / l2[r]; }
  float* Aout = A + base;
#pragma unroll
  for (int ot = 0; ot < 8; ot++) {
    const int dcol = ot * 16 + n16;
#pragma unroll
    for (int r = 0; r < 8; r++) {
      const int qrow = q0 + w * 16 + r + 8 * lhalf;
      Aout[(size_t)qrow * D_ + dcol] = O1[ot][r] * inv1[r] - O2[ot][r] * inv2[r];
    }
  }
}

// ---------------------------------------------------------------------------
// Kernel 3: group-norm stats. Group (b,g): all h, s in [g*128,(g+1)*128), d.
// Grid: 32 blocks of 256 threads.
// ---------------------------------------------------------------------------
__global__ void gn_stats(const float* __restrict__ A, float* __restrict__ mu,
                         float* __restrict__ rstd) {
  const int bg = blockIdx.x;
  const int b = bg >> 3, g = bg & 7;
  float s = 0.0f, q = 0.0f;
  for (int i = threadIdx.x; i < 8 * 128 * 128; i += 256) {
    const int h = i >> 14;
    const int rem = i & 16383;
    const int srow = (g << 7) + (rem >> 7);
    const int d = rem & 127;
    const float v = A[((size_t)(b * 8 + h) * 1024 + srow) * 128 + d];
    s += v; q += v * v;
  }
  __shared__ float ss[256], sq[256];
  ss[threadIdx.x] = s; sq[threadIdx.x] = q;
  __syncthreads();
  for (int st = 128; st > 0; st >>= 1) {
    if (threadIdx.x < st) {
      ss[threadIdx.x] += ss[threadIdx.x + st];
      sq[threadIdx.x] += sq[threadIdx.x + st];
    }
    __syncthreads();
  }
  if (threadIdx.x == 0) {
    const float m = ss[0] * (1.0f / 131072.0f);
    const float v = sq[0] * (1.0f / 131072.0f) - m * m;
    mu[bg] = m;
    rstd[bg] = rsqrtf(v + 1e-5f);
  }
}

// ---------------------------------------------------------------------------
// Kernel 4: out = Yhat @ Wo^T + bo with the reshape gather + group norm fused
// into the A-fragment load.  Y[b,s3,c] = Ahat[b, s3%8, (c/128)*128 + s3/8, c%128]
// Grid: (4096/64, 128/64), block 128.
// ---------------------------------------------------------------------------
__global__ __launch_bounds__(128, 1)
void out_proj_kernel(const float* __restrict__ A, const float* __restrict__ mu,
                     const float* __restrict__ rstd,
                     const float* __restrict__ gn_w,
                     const float* __restrict__ gn_b,
                     const float* __restrict__ lam_initP,
                     const float* __restrict__ Wo, const float* __restrict__ bo,
                     float* __restrict__ out) {
  const int tid = threadIdx.x, lane = tid & 31, w = tid >> 5;
  const int lhalf = lane >> 4, n16 = lane & 15;
  const int koffA = lhalf ? 16 : 0, koffB = lhalf ? 16 : 0;
  const int mtile = blockIdx.x * 64, ntile = blockIdx.y * 64;
  const float oneml = 1.0f - lam_initP[0];

  const int mrow = mtile + w * 16 + n16;  // this lane's A-operand row
  const int b = mrow >> 10, s3 = mrow & 1023;
  const int hh = s3 & 7, soff = s3 >> 3;

  v8f acc[4];
#pragma unroll
  for (int nt = 0; nt < 4; nt++) acc[nt] = v8f{};

  for (int ks = 0; ks < 32; ks++) {          // K = 1024 in steps of 32
    const int g = ks >> 2;                   // 128-wide c-block -> group id
    const int d0 = (ks & 3) * 32;
    const float* Ar =
        A + ((size_t)(b * 8 + hh) * 1024 + (g * 128 + soff)) * 128;
    const float mval = mu[b * 8 + g];
    const float rs = rstd[b * 8 + g];
    const float scl = rs * gn_w[g] * oneml;
    const float bia = (gn_b[g] - mval * rs * gn_w[g]) * oneml;
    const float* p = Ar + d0 + koffA;
    v16h af;
#pragma unroll
    for (int i = 0; i < 8; i++) {
      af[i]     = (f16)(p[i] * scl + bia);
      af[i + 8] = (f16)(p[8 + i] * scl + bia);
    }
#pragma unroll
    for (int nt = 0; nt < 4; nt++) {
      const int ocol = ntile + nt * 16 + n16;
      const float* wr = Wo + (size_t)ocol * 1024 + ks * 32 + koffB;
      v16h bf;
#pragma unroll
      for (int i = 0; i < 16; i++) bf[i] = (f16)wr[i];
      acc[nt] = wmma_f16(af, bf, acc[nt]);
    }
  }
#pragma unroll
  for (int nt = 0; nt < 4; nt++) {
    const int ocol = ntile + nt * 16 + n16;
    const float bv = bo[ocol];
#pragma unroll
    for (int r = 0; r < 8; r++) {
      const int m = mtile + w * 16 + r + 8 * lhalf;
      out[(size_t)m * 128 + ocol] = acc[nt][r] + bv;
    }
  }
}

// ---------------------------------------------------------------------------
extern "C" void kernel_launch(void* const* d_in, const int* in_sizes, int n_in,
                              void* d_out, int out_size, void* d_ws,
                              size_t ws_size, hipStream_t stream) {
  const float* query    = (const float*)d_in[0];
  const float* key      = (const float*)d_in[1];
  const float* values   = (const float*)d_in[2];
  const float* Wq       = (const float*)d_in[3];
  const float* bq       = (const float*)d_in[4];
  const float* Wk       = (const float*)d_in[5];
  const float* bk       = (const float*)d_in[6];
  const float* Wv       = (const float*)d_in[7];
  const float* bv       = (const float*)d_in[8];
  const float* Wo       = (const float*)d_in[9];
  const float* bo       = (const float*)d_in[10];
  const float* gn_w     = (const float*)d_in[11];
  const float* gn_b     = (const float*)d_in[12];
  const float* lam      = (const float*)d_in[13];
  const float* lam_init = (const float*)d_in[14];
  float* out = (float*)d_out;

  // Workspace layout (per-head tensors, E = B*H*S*D = 4M elements)
  const size_t E  = (size_t)B_ * H_ * S_ * D_;   // 4194304
  const size_t FB = E * sizeof(f16);             // 8 MB
  char* wsp = (char*)d_ws;
  f16* Q1 = (f16*)(wsp + 0 * FB);
  f16* Q2 = (f16*)(wsp + 1 * FB);
  f16* K1 = (f16*)(wsp + 2 * FB);
  f16* K2 = (f16*)(wsp + 3 * FB);
  f16* Vt = (f16*)(wsp + 4 * FB);
  float* Abuf = (float*)(wsp + 5 * FB);          // 16 MB f32
  float* muB  = (float*)(wsp + 5 * FB + E * sizeof(float));
  float* rsB  = muB + 32;

  dim3 blk(128);
  proj_kernel<<<dim3(64, 32), blk, 0, stream>>>(query,  Wq, bq, Q1, Q2, 0);
  proj_kernel<<<dim3(64, 32), blk, 0, stream>>>(key,    Wk, bk, K1, K2, 0);
  proj_kernel<<<dim3(64, 16), blk, 0, stream>>>(values, Wv, bv, Vt, nullptr, 1);
  attn_kernel<<<dim3(S_ / 64, B_ * H_), blk, 0, stream>>>(Q1, Q2, K1, K2, Vt,
                                                          lam, Abuf);
  gn_stats<<<dim3(32), dim3(256), 0, stream>>>(Abuf, muB, rsB);
  out_proj_kernel<<<dim3(64, 2), blk, 0, stream>>>(Abuf, muB, rsB, gn_w, gn_b,
                                                   lam_init, Wo, bo, out);
}
